// GraphGPT_37177236914632
// MI455X (gfx1250) — matrix operand order repaired and checked by reading, not compiled, not run on hardware
//
#include <hip/hip_runtime.h>
#include <hip/hip_bf16.h>
#include <math.h>

// ---------------------------------------------------------------------------
// GPT forward for gfx1250 (MI455X): bf16 WMMA GEMMs + flash attention.
// B=8 T=2048 D=256 H=4 DH=64 L=4
// GEMMs use 2x4 register blocking: each wave computes 32 rows x 64 cols,
// 8 v_wmma_f32_16x16x32_bf16 per 32-deep k-step.
// ---------------------------------------------------------------------------

typedef __attribute__((ext_vector_type(8)))  __bf16 v8bf;
typedef __attribute__((ext_vector_type(16))) __bf16 v16bf;
typedef __attribute__((ext_vector_type(8)))  float  v8f;

#define NB 8            // batch
#define NT 2048         // seq len
#define ND 256          // model dim
#define NH 4            // heads
#define NDH 64          // head dim
#define NL 4            // layers
#define NM (NB * NT)    // rows = 16384

__device__ __forceinline__ unsigned short f2bf(float f) {
    union { float f; unsigned u; } v; v.f = f;
    unsigned r = v.u + 0x7FFFu + ((v.u >> 16) & 1u);
    return (unsigned short)(r >> 16);
}

// A-fragment (16xK tile, 16-bit): lane = row (L&15); elements 0..7 at p,
// elements 8..15 at p+16 (K-halves interleave per lane-half, ISA 7.12.2).
__device__ __forceinline__ v16bf lda_frag(const __bf16* p) {
    v8bf lo = *(const v8bf*)p;
    v8bf hi = *(const v8bf*)(p + 16);
    return __builtin_shufflevector(lo, hi, 0,1,2,3,4,5,6,7,8,9,10,11,12,13,14,15);
}

// B-fragment (Kx16 tile, 16-bit): lane = col (L&15); elements 0..7 at p,
// elements 8..15 at p+16, p offset by (lane>>4)*8 (same K order as A).
__device__ __forceinline__ v16bf ldb_frag(const __bf16* p) {
    v8bf lo = *(const v8bf*)p;
    v8bf hi = *(const v8bf*)(p + 16);
    return __builtin_shufflevector(lo, hi, 0,1,2,3,4,5,6,7,8,9,10,11,12,13,14,15);
}

// Register-blocked GEMM core: acc[2][4] covers rows [row0,row0+32) x
// cols [col0,col0+64). A row-major [M,K] bf16; Bt transposed [N,K] bf16.
__device__ __forceinline__ void gemm_block(const __bf16* __restrict__ A,
                                           const __bf16* __restrict__ Bt,
                                           int K, int row0, int col0,
                                           int half, int r, v8f acc[2][4]) {
    const __bf16* ap0 = A + (size_t)(row0 + r) * K + half * 8;
    const __bf16* ap1 = ap0 + (size_t)16 * K;
    const __bf16* bp  = Bt + (size_t)(col0 + r) * K + half * 8;
    const size_t bstride = (size_t)16 * K;
    for (int k0 = 0; k0 < K; k0 += 32) {
        v16bf a0 = lda_frag(ap0 + k0);
        v16bf a1 = lda_frag(ap1 + k0);
        v16bf b0 = ldb_frag(bp + k0);
        v16bf b1 = ldb_frag(bp + bstride + k0);
        v16bf b2 = ldb_frag(bp + 2 * bstride + k0);
        v16bf b3 = ldb_frag(bp + 3 * bstride + k0);
        acc[0][0] = __builtin_amdgcn_wmma_f32_16x16x32_bf16(false, a0, false, b0, (short)0, acc[0][0], false, false);
        acc[0][1] = __builtin_amdgcn_wmma_f32_16x16x32_bf16(false, a0, false, b1, (short)0, acc[0][1], false, false);
        acc[0][2] = __builtin_amdgcn_wmma_f32_16x16x32_bf16(false, a0, false, b2, (short)0, acc[0][2], false, false);
        acc[0][3] = __builtin_amdgcn_wmma_f32_16x16x32_bf16(false, a0, false, b3, (short)0, acc[0][3], false, false);
        acc[1][0] = __builtin_amdgcn_wmma_f32_16x16x32_bf16(false, a1, false, b0, (short)0, acc[1][0], false, false);
        acc[1][1] = __builtin_amdgcn_wmma_f32_16x16x32_bf16(false, a1, false, b1, (short)0, acc[1][1], false, false);
        acc[1][2] = __builtin_amdgcn_wmma_f32_16x16x32_bf16(false, a1, false, b2, (short)0, acc[1][2], false, false);
        acc[1][3] = __builtin_amdgcn_wmma_f32_16x16x32_bf16(false, a1, false, b3, (short)0, acc[1][3], false, false);
    }
}

__device__ __forceinline__ void zero_acc(v8f acc[2][4]) {
    const v8f vz = {0.f,0.f,0.f,0.f,0.f,0.f,0.f,0.f};
#pragma unroll
    for (int t = 0; t < 2; ++t)
#pragma unroll
        for (int j = 0; j < 4; ++j) acc[t][j] = vz;
}

// ---------------------------------------------------------------------------
// Elementwise / setup kernels
// ---------------------------------------------------------------------------

__global__ __launch_bounds__(256) void embed_kernel(const int* __restrict__ ids,
                                                    const float* __restrict__ tok,
                                                    const float* __restrict__ pos,
                                                    float* __restrict__ X) {
    int i = blockIdx.x * 256 + threadIdx.x;           // over NM*ND
    int d = i & (ND - 1);
    int m = i >> 8;
    int t = m & (NT - 1);
    X[i] = tok[(size_t)ids[m] * ND + d] + pos[(size_t)t * ND + d];
}

// f32 [K,N] -> bf16 transposed [N,K]
__global__ __launch_bounds__(256) void convert_w_kernel(const float* __restrict__ src,
                                                        unsigned short* __restrict__ dst,
                                                        int K, int N) {
    int i = blockIdx.x * 256 + threadIdx.x;           // over K*N
    int k = i / N, n = i - k * N;
    dst[(size_t)n * K + k] = f2bf(src[i]);
}

// LayerNorm over D=256, one row per 256-thread block.
__global__ __launch_bounds__(256) void ln_bf16_kernel(const float* __restrict__ X,
                                                      const float* __restrict__ g,
                                                      const float* __restrict__ b,
                                                      unsigned short* __restrict__ Y) {
    __shared__ float red[256];
    int row = blockIdx.x, tid = threadIdx.x;
    float v = X[(size_t)row * ND + tid];
    red[tid] = v; __syncthreads();
    for (int s = 128; s > 0; s >>= 1) { if (tid < s) red[tid] += red[tid + s]; __syncthreads(); }
    float mean = red[0] * (1.f / ND);
    __syncthreads();
    float d = v - mean;
    red[tid] = d * d; __syncthreads();
    for (int s = 128; s > 0; s >>= 1) { if (tid < s) red[tid] += red[tid + s]; __syncthreads(); }
    float var = red[0] * (1.f / ND);
    float y = d * rsqrtf(var + 1e-5f) * g[tid] + b[tid];
    Y[(size_t)row * ND + tid] = f2bf(y);
}

__global__ __launch_bounds__(256) void ln_f32_kernel(const float* __restrict__ X,
                                                     const float* __restrict__ g,
                                                     const float* __restrict__ b,
                                                     float* __restrict__ Y) {
    __shared__ float red[256];
    int row = blockIdx.x, tid = threadIdx.x;
    float v = X[(size_t)row * ND + tid];
    red[tid] = v; __syncthreads();
    for (int s = 128; s > 0; s >>= 1) { if (tid < s) red[tid] += red[tid + s]; __syncthreads(); }
    float mean = red[0] * (1.f / ND);
    __syncthreads();
    float d = v - mean;
    red[tid] = d * d; __syncthreads();
    for (int s = 128; s > 0; s >>= 1) { if (tid < s) red[tid] += red[tid + s]; __syncthreads(); }
    float var = red[0] * (1.f / ND);
    Y[(size_t)row * ND + tid] = d * rsqrtf(var + 1e-5f) * g[tid] + b[tid];
}

// ---------------------------------------------------------------------------
// GEMM kernels: 256 threads = 8 waves arranged 2(M) x 4(N); block tile is
// 64 rows x 256 cols; each wave owns 32x64 via acc[2][4].
// ---------------------------------------------------------------------------

// QKV projection: Y[M,256] @ Wqkv^T[768,256] + bias -> scatter into Q/K/Vt bf16.
__global__ __launch_bounds__(256) void gemm_qkv_kernel(const unsigned short* __restrict__ Yb,
                                                       const unsigned short* __restrict__ Wt,
                                                       const float* __restrict__ bias,
                                                       unsigned short* __restrict__ Qb,
                                                       unsigned short* __restrict__ Kb,
                                                       unsigned short* __restrict__ Vtb) {
    int lane = threadIdx.x & 31, wv = threadIdx.x >> 5;
    int half = lane >> 4, r = lane & 15;
    int row0 = blockIdx.y * 64 + (wv >> 2) * 32;
    int col0 = blockIdx.x * 256 + (wv & 3) * 64;
    v8f acc[2][4];
    zero_acc(acc);
    gemm_block((const __bf16*)Yb, (const __bf16*)Wt, ND, row0, col0, half, r, acc);

#pragma unroll
    for (int j = 0; j < 4; ++j) {
        int c  = col0 + 16 * j + r;        // 0..767
        float bv = bias[c];
        int s  = c >> 8;                   // 0=q 1=k 2=v
        int h  = (c >> 6) & 3;
        int dh = c & 63;
#pragma unroll
        for (int t = 0; t < 2; ++t) {
#pragma unroll
            for (int i = 0; i < 8; ++i) {
                int m = row0 + 16 * t + half * 8 + i;
                int bb = m >> 11, tt = m & (NT - 1);
                float val = acc[t][j][i] + bv;
                size_t hb = (size_t)(bb * NH + h);
                if (s == 0)      Qb[(hb * NT + tt) * NDH + dh] = f2bf(val * 0.125f);  // fold 1/sqrt(64)
                else if (s == 1) Kb[(hb * NT + tt) * NDH + dh] = f2bf(val);
                else             Vtb[(hb * NDH + dh) * NT + tt] = f2bf(val);
            }
        }
    }
}

// Out-proj / fc2: A[M,K]bf16 @ Wt[N,K] + bias + residual -> X f32 (in place).
__global__ __launch_bounds__(256) void gemm_resid_kernel(const unsigned short* __restrict__ Ab,
                                                         const unsigned short* __restrict__ Wt,
                                                         const float* __restrict__ bias,
                                                         float* __restrict__ X,
                                                         int N, int K) {
    int lane = threadIdx.x & 31, wv = threadIdx.x >> 5;
    int half = lane >> 4, r = lane & 15;
    int row0 = blockIdx.y * 64 + (wv >> 2) * 32;
    int col0 = blockIdx.x * 256 + (wv & 3) * 64;
    v8f acc[2][4];
    zero_acc(acc);
    gemm_block((const __bf16*)Ab, (const __bf16*)Wt, K, row0, col0, half, r, acc);
#pragma unroll
    for (int j = 0; j < 4; ++j) {
        float bv = bias[col0 + 16 * j + r];
#pragma unroll
        for (int t = 0; t < 2; ++t) {
#pragma unroll
            for (int i = 0; i < 8; ++i) {
                size_t idx = (size_t)(row0 + 16 * t + half * 8 + i) * N + col0 + 16 * j + r;
                X[idx] = X[idx] + acc[t][j][i] + bv;
            }
        }
    }
}

// fc1: A[M,256] @ Wt[1024,256] + bias, exact GELU, -> bf16.
__global__ __launch_bounds__(256) void gemm_gelu_kernel(const unsigned short* __restrict__ Ab,
                                                        const unsigned short* __restrict__ Wt,
                                                        const float* __restrict__ bias,
                                                        unsigned short* __restrict__ Hb) {
    int lane = threadIdx.x & 31, wv = threadIdx.x >> 5;
    int half = lane >> 4, r = lane & 15;
    int row0 = blockIdx.y * 64 + (wv >> 2) * 32;
    int col0 = blockIdx.x * 256 + (wv & 3) * 64;
    v8f acc[2][4];
    zero_acc(acc);
    gemm_block((const __bf16*)Ab, (const __bf16*)Wt, ND, row0, col0, half, r, acc);
#pragma unroll
    for (int j = 0; j < 4; ++j) {
        float bv = bias[col0 + 16 * j + r];
#pragma unroll
        for (int t = 0; t < 2; ++t) {
#pragma unroll
            for (int i = 0; i < 8; ++i) {
                float u = acc[t][j][i] + bv;
                float ge = 0.5f * u * (1.f + erff(u * 0.70710678118654752f));
                Hb[(size_t)(row0 + 16 * t + half * 8 + i) * (4 * ND) + col0 + 16 * j + r] = f2bf(ge);
            }
        }
    }
}

// ---------------------------------------------------------------------------
// Flash attention: grid (T/64, B*H), 4 waves/block, each wave = 16 queries.
// Q,K: [B,H,T,DH] bf16 (Q pre-scaled). Vt: [B,H,DH,T] bf16.
// Writes head_out f32 [B,H,T,DH] and attn-concat bf16 [B,T,D].
// ---------------------------------------------------------------------------
__device__ __forceinline__ float redmax16(float v) {
    v = fmaxf(v, __shfl_xor(v, 1, 32));
    v = fmaxf(v, __shfl_xor(v, 2, 32));
    v = fmaxf(v, __shfl_xor(v, 4, 32));
    v = fmaxf(v, __shfl_xor(v, 8, 32));
    return v;
}
__device__ __forceinline__ float redsum16(float v) {
    v += __shfl_xor(v, 1, 32);
    v += __shfl_xor(v, 2, 32);
    v += __shfl_xor(v, 4, 32);
    v += __shfl_xor(v, 8, 32);
    return v;
}

__global__ __launch_bounds__(128) void attn_kernel(const unsigned short* __restrict__ Qb,
                                                   const unsigned short* __restrict__ Kb,
                                                   const unsigned short* __restrict__ Vtb,
                                                   unsigned short* __restrict__ attnc,
                                                   float* __restrict__ head_out) {
    __shared__ unsigned short plds[4][16 * 40];      // per-wave 16x32 P tile, stride 40
    int lane = threadIdx.x & 31, wv = threadIdx.x >> 5;
    int half = lane >> 4, r = lane & 15;
    int q0 = blockIdx.x * 64 + wv * 16;
    int bh = blockIdx.y;
    const __bf16* Q = (const __bf16*)Qb + (size_t)bh * NT * NDH;
    const __bf16* K = (const __bf16*)Kb + (size_t)bh * NT * NDH;
    const __bf16* V = (const __bf16*)Vtb + (size_t)bh * NDH * NT;
    unsigned short* pl = plds[wv];

    // Q fragments for dh 0..31 and 32..63 (loaded once)
    const __bf16* qp = Q + (size_t)(q0 + r) * NDH + half * 8;
    v16bf qa0 = lda_frag(qp);
    v16bf qa1 = lda_frag(qp + 32);

    const v8f vz = {0.f,0.f,0.f,0.f,0.f,0.f,0.f,0.f};
    v8f o[4]; o[0] = vz; o[1] = vz; o[2] = vz; o[3] = vz;
    float mrow[8], lrow[8];
#pragma unroll
    for (int i = 0; i < 8; ++i) { mrow[i] = -1e30f; lrow[i] = 0.f; }

    for (int key0 = 0; key0 < q0 + 16; key0 += 32) {
        // --- scores: S = Q K^T for 16q x 32k (two 16x16 tiles) ---
        const __bf16* kp0 = K + (size_t)(key0 + r) * NDH + half * 8;
        const __bf16* kp1 = K + (size_t)(key0 + 16 + r) * NDH + half * 8;
        v8f s0 = __builtin_amdgcn_wmma_f32_16x16x32_bf16(false, qa0, false, ldb_frag(kp0),      (short)0, vz, false, false);
        s0     = __builtin_amdgcn_wmma_f32_16x16x32_bf16(false, qa1, false, ldb_frag(kp0 + 32), (short)0, s0, false, false);
        v8f s1 = __builtin_amdgcn_wmma_f32_16x16x32_bf16(false, qa0, false, ldb_frag(kp1),      (short)0, vz, false, false);
        s1     = __builtin_amdgcn_wmma_f32_16x16x32_bf16(false, qa1, false, ldb_frag(kp1 + 32), (short)0, s1, false, false);

        // --- causal mask + online softmax update ---
        float p0[8], p1[8];
#pragma unroll
        for (int i = 0; i < 8; ++i) {
            int q = q0 + half * 8 + i;
            float a = (key0 + r      > q) ? -1e30f : s0[i];
            float b = (key0 + 16 + r > q) ? -1e30f : s1[i];
            float rm = redmax16(fmaxf(a, b));
            float mn = fmaxf(mrow[i], rm);
            float alpha = __expf(mrow[i] - mn);
            float e0 = __expf(a - mn);
            float e1 = __expf(b - mn);
            lrow[i] = lrow[i] * alpha + redsum16(e0 + e1);
            mrow[i] = mn;
            o[0][i] *= alpha; o[1][i] *= alpha; o[2][i] *= alpha; o[3][i] *= alpha;
            p0[i] = e0; p1[i] = e1;
        }

        // --- transpose P (C-layout) -> A-fragment via per-wave LDS ---
#pragma unroll
        for (int i = 0; i < 8; ++i) {
            int row = half * 8 + i;
            pl[row * 40 + r]      = f2bf(p0[i]);
            pl[row * 40 + 16 + r] = f2bf(p1[i]);
        }
        asm volatile("s_wait_dscnt 0" ::: "memory");
        v16bf pfrag = lda_frag((const __bf16*)(pl + r * 40 + half * 8));

        // --- O += P V (4 column tiles of DH) ---
#pragma unroll
        for (int j = 0; j < 4; ++j) {
            const __bf16* vp = V + (size_t)(16 * j + r) * NT + key0 + half * 8;
            o[j] = __builtin_amdgcn_wmma_f32_16x16x32_bf16(false, pfrag, false, ldb_frag(vp),
                                                           (short)0, o[j], false, false);
        }
    }

    // --- epilogue: normalize, write head_out f32 + concat bf16 ---
    int bb = bh >> 2, h = bh & 3;
#pragma unroll
    for (int i = 0; i < 8; ++i) {
        int q = q0 + half * 8 + i;
        float inv = 1.f / lrow[i];
#pragma unroll
        for (int j = 0; j < 4; ++j) {
            int dh = 16 * j + r;
            float val = o[j][i] * inv;
            head_out[(size_t)bh * NT * NDH + (size_t)q * NDH + dh] = val;
            attnc[((size_t)(bb * NT + q)) * ND + h * NDH + dh] = f2bf(val);
        }
    }
}

// ---------------------------------------------------------------------------
// Host-side orchestration
// ---------------------------------------------------------------------------
extern "C" void kernel_launch(void* const* d_in, const int* in_sizes, int n_in,
                              void* d_out, int out_size, void* d_ws, size_t ws_size,
                              hipStream_t stream) {
    (void)in_sizes; (void)n_in; (void)out_size; (void)ws_size;
    const int*   ids   = (const int*)  d_in[0];
    const float* tok   = (const float*)d_in[2];
    const float* pos   = (const float*)d_in[3];
    const float* qkv_w = (const float*)d_in[4];
    const float* qkv_b = (const float*)d_in[5];
    const float* out_w = (const float*)d_in[6];
    const float* out_b = (const float*)d_in[7];
    const float* ln1_g = (const float*)d_in[8];
    const float* ln1_b = (const float*)d_in[9];
    const float* ln2_g = (const float*)d_in[10];
    const float* ln2_b = (const float*)d_in[11];
    const float* fc1_w = (const float*)d_in[12];
    const float* fc1_b = (const float*)d_in[13];
    const float* fc2_w = (const float*)d_in[14];
    const float* fc2_b = (const float*)d_in[15];
    const float* lnf_g = (const float*)d_in[16];
    const float* lnf_b = (const float*)d_in[17];

    char* w = (char*)d_ws;
    float*          X   = (float*)         (w + 0);          // 16 MB  [M,D] f32
    unsigned short* Y   = (unsigned short*)(w + 16777216);   //  8 MB  [M,D] bf16 (LN out)
    unsigned short* Qb  = (unsigned short*)(w + 25165824);   //  8 MB
    unsigned short* Kb  = (unsigned short*)(w + 33554432);   //  8 MB
    unsigned short* Vtb = (unsigned short*)(w + 41943040);   //  8 MB
    unsigned short* ATT = (unsigned short*)(w + 50331648);   //  8 MB  [M,D] bf16
    unsigned short* Hb  = (unsigned short*)(w + 58720256);   // 32 MB  [M,4D] bf16
    unsigned short* Wq  = (unsigned short*)(w + 92274688);   // [768,256]
    unsigned short* Wo  = (unsigned short*)(w + 92667904);   // [256,256]
    unsigned short* W1  = (unsigned short*)(w + 92798976);   // [1024,256]
    unsigned short* W2  = (unsigned short*)(w + 93323264);   // [256,1024]

    float* hidden_out = (float*)d_out;                       // [B,T,D]
    float* head_out   = (float*)d_out + (size_t)NM * ND;     // [B,H,T,DH]

    // x = token_emb[ids] + pos_emb
    embed_kernel<<<NM * ND / 256, 256, 0, stream>>>(ids, tok, pos, X);

    for (int l = 0; l < NL; ++l) {
        const float* qW = qkv_w + (size_t)l * ND * 3 * ND;
        const float* oW = out_w + (size_t)l * ND * ND;
        const float* f1 = fc1_w + (size_t)l * ND * 4 * ND;
        const float* f2 = fc2_w + (size_t)l * 4 * ND * ND;

        convert_w_kernel<<<ND * 3 * ND / 256, 256, 0, stream>>>(qW, Wq, ND, 3 * ND);
        convert_w_kernel<<<ND * ND / 256,     256, 0, stream>>>(oW, Wo, ND, ND);
        convert_w_kernel<<<ND * 4 * ND / 256, 256, 0, stream>>>(f1, W1, ND, 4 * ND);
        convert_w_kernel<<<4 * ND * ND / 256, 256, 0, stream>>>(f2, W2, 4 * ND, ND);

        // ln1 -> Y (bf16)
        ln_bf16_kernel<<<NM, 256, 0, stream>>>(X, ln1_g + l * ND, ln1_b + l * ND, Y);
        // QKV projection + split/scale/transpose-V  (N=768 -> 3 col-blocks)
        gemm_qkv_kernel<<<dim3(3, NM / 64), 256, 0, stream>>>(Y, Wq, qkv_b + (size_t)l * 3 * ND,
                                                              Qb, Kb, Vtb);
        // flash attention -> ATT (bf16 concat) + head_out (f32)
        attn_kernel<<<dim3(NT / 64, NB * NH), 128, 0, stream>>>(Qb, Kb, Vtb, ATT, head_out);
        // x += ATT @ Wo + bo   (N=256 -> 1 col-block)
        gemm_resid_kernel<<<dim3(1, NM / 64), 256, 0, stream>>>(ATT, Wo, out_b + (size_t)l * ND,
                                                                X, ND, ND);
        // ln2 -> Y
        ln_bf16_kernel<<<NM, 256, 0, stream>>>(X, ln2_g + l * ND, ln2_b + l * ND, Y);
        // h = gelu(Y @ W1 + b1)  (N=1024 -> 4 col-blocks)
        gemm_gelu_kernel<<<dim3(4, NM / 64), 256, 0, stream>>>(Y, W1, fc1_b + (size_t)l * 4 * ND, Hb);
        // x += h @ W2 + b2   (N=256, K=1024)
        gemm_resid_kernel<<<dim3(1, NM / 64), 256, 0, stream>>>(Hb, W2, fc2_b + (size_t)l * ND,
                                                                X, ND, 4 * ND);
    }

    // final layernorm -> d_out part 1
    ln_f32_kernel<<<NM, 256, 0, stream>>>(X, lnf_g, lnf_b, hidden_out);
}
